// VanillaVAE_47304769798545
// MI455X (gfx1250) — matrix operand order repaired and checked by reading, not compiled, not run on hardware
//
#include <hip/hip_runtime.h>

// ---------------------------------------------------------------------------
// GRU-VAE fused persistent kernel for gfx1250 (MI455X), round 4.
// - BT=16 batch rows per workgroup, ping-pong hidden buffers in LDS
// - recurrent GEMMs and decoder head on v_wmma_f32_16x16x32_f16
// - layer-1 r/z gates: input+recurrent GEMMs share one accumulator
// - biases ride in the WMMA accumulator init (splat into C)
// - h_prev carried in registers; token indices pre-scaled for addressing
// - hardware transcendentals (v_exp/v_rcp/v_tanh)
// ---------------------------------------------------------------------------

namespace {
constexpr int Bb = 2048, Ss = 256, Vv = 4, Ee = 8, Hh = 64, LATn = 3, NTt = 6;
constexpr int START_TOK = 4;
constexpr int G3 = 3 * Hh;              // 192 gate columns
constexpr int BT = 16;                  // batch rows per workgroup
constexpr int THREADS = 128;            // 4 wave32
constexpr int HSTR = 72;                // padded LDS row stride (halves)
constexpr int HBUF = BT * HSTR;         // one hidden buffer (halves)
constexpr int TABF = NTt * G3;          // 1152 floats per gi-table
constexpr int FRAGW = 12 * 2 * 32 * 16; // 12288 halves per swizzled weight
constexpr int FRAGH = 2 * 32 * 16;      // 1024 halves for the padded head tile
// d_out layout (flat, return order: recons, tokens, mu, log_var)
constexpr int REC_SZ  = Bb * Ss * Vv;
constexpr int TOK_OFF = REC_SZ;
constexpr int MU_OFF  = REC_SZ + Bb * Ss;
constexpr int LV_OFF  = MU_OFF + Bb * LATn;
static_assert((Ss & 1) == 0, "ping-pong assumes even S");
}

typedef _Float16 h16v __attribute__((ext_vector_type(16)));
typedef _Float16 h8v  __attribute__((ext_vector_type(8)));
typedef float    f8v  __attribute__((ext_vector_type(8)));

// ------------------------- fast transcendentals ----------------------------

__device__ __forceinline__ float fexp(float x) {
#if __has_builtin(__builtin_amdgcn_exp2f)
  return __builtin_amdgcn_exp2f(x * 1.44269504f);
#else
  return __expf(x);
#endif
}
__device__ __forceinline__ float frcp(float x) {
#if __has_builtin(__builtin_amdgcn_rcpf)
  return __builtin_amdgcn_rcpf(x);
#else
  return 1.f / x;
#endif
}
__device__ __forceinline__ float fsig(float x) { return frcp(1.f + fexp(-x)); }
__device__ __forceinline__ float ftanh(float x) {
#if __has_builtin(__builtin_amdgcn_tanhf)
  return __builtin_amdgcn_tanhf(x);
#else
  float e = fexp(2.f * x);
  return (e - 1.f) * frcp(e + 1.f);
#endif
}

__device__ __forceinline__ unsigned hashu(unsigned x) {
  x ^= x >> 16; x *= 0x7feb352dU; x ^= x >> 15; x *= 0x846ca68bU; x ^= x >> 16;
  return x;
}
__device__ __forceinline__ float rnd01(unsigned a, unsigned b) {
  return (hashu(a * 0x9E3779B1u + b * 0x85EBCA77u + 0x165667B1u) >> 8)
         * (1.0f / 16777216.0f);
}

__device__ __forceinline__ f8v splat8(float x) {
  f8v v = {x, x, x, x, x, x, x, x};
  return v;
}

// ------------------------------ WMMA helpers -------------------------------

__device__ __forceinline__ f8v wmma_f16(h16v a, h16v b, f8v c) {
  return __builtin_amdgcn_wmma_f32_16x16x32_f16(false, a, false, b,
                                                (short)0, c, false, false);
}

__device__ __forceinline__ h16v bfrag(const _Float16* __restrict__ wb,
                                      int j, int c, int lane) {
  return *(const h16v*)(wb + (((j * 2 + c) * 32) + lane) * 16);
}

// A-fragment (16x32 f16) from LDS hidden buffer (documented CDNA5 A layout).
__device__ __forceinline__ h16v load_afrag(const _Float16* hbuf, int lane, int c) {
  int row = lane & 15;
  int kb  = c * 32 + ((lane >> 4) << 3);
  const _Float16* p = hbuf + row * HSTR + kb;
  h8v lo = *(const h8v*)(p);
  h8v hi = *(const h8v*)(p + 16);
  h16v r;
#pragma unroll
  for (int i = 0; i < 8; ++i) { r[i] = lo[i]; r[i + 8] = hi[i]; }
  return r;
}

// 3 gate-aligned N-tiles of one [16,64]@[64,192] GEMM (layer-0 recurrent).
__device__ __forceinline__ void gemm3(f8v* acc, h16v a0, h16v a1,
                                      const _Float16* __restrict__ wb,
                                      int ng, int lane) {
#pragma unroll
  for (int g = 0; g < 3; ++g) {
    int j = ng + g * 4;
    acc[g] = wmma_f16(a0, bfrag(wb, j, 0, lane), acc[g]);
    acc[g] = wmma_f16(a1, bfrag(wb, j, 1, lane), acc[g]);
  }
}

// One N-tile accumulating BOTH input (x@Wi) and recurrent (h@Wh) GEMMs.
__device__ __forceinline__ f8v gemm_pair(f8v acc, h16v x0, h16v x1,
                                         h16v y0, h16v y1,
                                         const _Float16* __restrict__ wi,
                                         const _Float16* __restrict__ wh,
                                         int j, int lane) {
  acc = wmma_f16(x0, bfrag(wi, j, 0, lane), acc);
  acc = wmma_f16(x1, bfrag(wi, j, 1, lane), acc);
  acc = wmma_f16(y0, bfrag(wh, j, 0, lane), acc);
  acc = wmma_f16(y1, bfrag(wh, j, 1, lane), acc);
  return acc;
}

// ------------------------------ prep kernels -------------------------------

// gi tables (token-indexed input projections); bhh0 r/z folded in.
__global__ void prep_tables(const float* __restrict__ emb,
                            const float* __restrict__ eWih0,
                            const float* __restrict__ ebih0,
                            const float* __restrict__ ebhh0,
                            const float* __restrict__ demb,
                            const float* __restrict__ dWih0,
                            const float* __restrict__ dbih0,
                            const float* __restrict__ dbhh0,
                            float* __restrict__ tabs) {
  int idx = blockIdx.x * 256 + threadIdx.x;
  if (idx < TABF) {
    int t = idx / G3, col = idx % G3;
    float v = ebih0[col] + (col < 2 * Hh ? ebhh0[col] : 0.f);
    for (int e = 0; e < Ee; ++e) v += emb[t * Ee + e] * eWih0[col * Ee + e];
    tabs[idx] = v;
  } else if (idx < 2 * TABF) {
    int i2 = idx - TABF;
    int t = i2 / G3, col = i2 % G3;
    float v = dbih0[col] + (col < 2 * Hh ? dbhh0[col] : 0.f);
    for (int k = 0; k < Hh; ++k) v += demb[t * Hh + k] * dWih0[col * Hh + k];
    tabs[idx] = v;
  }
}

// Swizzle six [192,64] weights + zero-padded [4->16,64] head into per-lane
// f16 B-fragments.
__global__ void prep_frags(const float* __restrict__ w0, const float* __restrict__ w1,
                           const float* __restrict__ w2, const float* __restrict__ w3,
                           const float* __restrict__ w4, const float* __restrict__ w5,
                           const float* __restrict__ headW,
                           _Float16* __restrict__ frags) {
  int idx = blockIdx.x * 256 + threadIdx.x;
  if (idx < 6 * FRAGW) {
    int w = idx / FRAGW, rem = idx % FRAGW;
    int j = rem >> 10, c = (rem >> 9) & 1, lane = (rem >> 4) & 31, i = rem & 15;
    int n = j * 16 + (lane & 15);
    int k = ((lane >> 4) << 3) + c * 32 + i + ((i >> 3) << 3);
    const float* W = (w == 0) ? w0 : (w == 1) ? w1 : (w == 2) ? w2
                    : (w == 3) ? w3 : (w == 4) ? w4 : w5;
    frags[idx] = (_Float16)W[n * Hh + k];
  } else if (idx < 6 * FRAGW + FRAGH) {
    int rem = idx - 6 * FRAGW;
    int c = rem >> 9, lane = (rem >> 4) & 31, i = rem & 15;
    int n = lane & 15;
    int k = ((lane >> 4) << 3) + c * 32 + i + ((i >> 3) << 3);
    frags[idx] = (n < Vv) ? (_Float16)headW[n * Hh + k] : (_Float16)0.f;
  }
}

__global__ void copy_tokens(const int* __restrict__ t, int* __restrict__ o) {
  int i = blockIdx.x * 256 + threadIdx.x;
  if (i < Bb * Ss) o[TOK_OFF + i] = t[i];
}

// ------------------------------ main kernel --------------------------------

__global__ __launch_bounds__(THREADS)
void vae_fused(const int* __restrict__ tokens,
               const float* __restrict__ fc_mu_W, const float* __restrict__ fc_mu_b,
               const float* __restrict__ fc_var_W, const float* __restrict__ fc_var_b,
               const float* __restrict__ proj_W, const float* __restrict__ proj_b,
               const float* __restrict__ head_b,
               const float* __restrict__ enc_bhh0, const float* __restrict__ enc_bih1,
               const float* __restrict__ enc_bhh1,
               const float* __restrict__ dec_bhh0, const float* __restrict__ dec_bih1,
               const float* __restrict__ dec_bhh1,
               const float* __restrict__ tabs, const _Float16* __restrict__ frags,
               float* __restrict__ out) {
  __shared__ _Float16 h0s[2 * HBUF];            // ping-pong layer-0 hidden
  __shared__ _Float16 h1s[2 * HBUF];            // ping-pong layer-1 hidden
  __shared__ float gte[TABF];                   // encoder gi table
  __shared__ float gtd[TABF];                   // decoder gi table
  __shared__ int   toksT[Ss * BT];              // tokens*G3, transposed [t][m]
  __shared__ __align__(16) int toks[BT];        // decoder sampled tokens * G3
  __shared__ __align__(32) float lgs[Vv * BT];  // head logits [v][row]

  const int tid   = threadIdx.x;
  const int lane  = tid & 31;
  const int ng    = tid >> 5;                // wave id = hidden-column group
  const int colh  = ng * 16 + (lane & 15);   // hidden column [0,64)
  const int mbase = (lane >> 4) << 3;        // C-layout row base (0 or 8)
  const int b0    = blockIdx.x * BT;

  // stage tables + pre-scaled transposed tokens, zero buffer 0, prefetch
  for (int i = tid; i < TABF; i += THREADS) { gte[i] = tabs[i]; gtd[i] = tabs[TABF + i]; }
  for (int i = tid; i < Ss * BT; i += THREADS)
    toksT[i] = tokens[(b0 + (i & (BT - 1))) * Ss + (i >> 4)] * G3;
  for (int i = tid; i < HBUF; i += THREADS) {
    h0s[i] = (_Float16)0.f; h1s[i] = (_Float16)0.f;
  }
  for (int i = tid; i < (6 * FRAGW + FRAGH) / 64; i += THREADS)
    __builtin_prefetch(frags + (size_t)i * 64, 0, 3);

  // loop-invariant bias scalars (splatted into WMMA accumulators below)
  const float be0g  = enc_bhh0[2 * Hh + colh];
  const float be1r  = enc_bih1[colh] + enc_bhh1[colh];
  const float be1z  = enc_bih1[Hh + colh] + enc_bhh1[Hh + colh];
  const float be1ig = enc_bih1[2 * Hh + colh];
  const float be1hg = enc_bhh1[2 * Hh + colh];
  const float bd0g  = dec_bhh0[2 * Hh + colh];
  const float bd1r  = dec_bih1[colh] + dec_bhh1[colh];
  const float bd1z  = dec_bih1[Hh + colh] + dec_bhh1[Hh + colh];
  const float bd1ig = dec_bih1[2 * Hh + colh];
  const float bd1hg = dec_bhh1[2 * Hh + colh];
  const float hbn   = ((lane & 15) < Vv) ? head_b[lane & 15] : 0.f;

  const _Float16* W_eh0 = frags + 0 * FRAGW;
  const _Float16* W_ei1 = frags + 1 * FRAGW;
  const _Float16* W_eh1 = frags + 2 * FRAGW;
  const _Float16* W_dh0 = frags + 3 * FRAGW;
  const _Float16* W_di1 = frags + 4 * FRAGW;
  const _Float16* W_dh1 = frags + 5 * FRAGW;
  const _Float16* W_hd  = frags + 6 * FRAGW;

  float hp0[8], hp1[8];                      // h_prev carried in registers
#pragma unroll
  for (int r = 0; r < 8; ++r) { hp0[r] = 0.f; hp1[r] = 0.f; }

  __syncthreads();

  // ------------------------------ encoder ------------------------------
  for (int t = 0; t < Ss; ++t) {
    const int cur = t & 1, nxt = cur ^ 1;
    const _Float16* h0c = h0s + cur * HBUF;
    const _Float16* h1c = h1s + cur * HBUF;
    _Float16* h0n = h0s + nxt * HBUF;
    _Float16* h1n = h1s + nxt * HBUF;

    h16v a0_0 = load_afrag(h0c, lane, 0);
    h16v a0_1 = load_afrag(h0c, lane, 1);
    h16v a1_0 = load_afrag(h1c, lane, 0);
    h16v a1_1 = load_afrag(h1c, lane, 1);
    int tks[8];
    *(int4*)&tks[0] = *(const int4*)&toksT[t * BT + mbase];
    *(int4*)&tks[4] = *(const int4*)&toksT[t * BT + mbase + 4];

    f8v gh0[3] = {splat8(0.f), splat8(0.f), splat8(be0g)};
    gemm3(gh0, a0_0, a0_1, W_eh0, ng, lane);
#pragma unroll
    for (int r = 0; r < 8; ++r) {
      int m = mbase + r;
      const float* gt = &gte[tks[r] + colh];
      float rr = fsig(gt[0] + gh0[0][r]);
      float zz = fsig(gt[Hh] + gh0[1][r]);
      float nn = ftanh(gt[2 * Hh] + rr * gh0[2][r]);
      float hn = (1.f - zz) * nn + zz * hp0[r];
      hp0[r] = hn;
      h0n[m * HSTR + colh] = (_Float16)hn;
    }
    __syncthreads();                              // layer-0 output committed

    h16v an0 = load_afrag(h0n, lane, 0);
    h16v an1 = load_afrag(h0n, lane, 1);
    // r/z gates: input + recurrent share one accumulator; n gate split
    f8v accr  = gemm_pair(splat8(be1r), an0, an1, a1_0, a1_1, W_ei1, W_eh1, ng, lane);
    f8v accz  = gemm_pair(splat8(be1z), an0, an1, a1_0, a1_1, W_ei1, W_eh1, ng + 4, lane);
    f8v accig = splat8(be1ig);
    accig = wmma_f16(an0, bfrag(W_ei1, ng + 8, 0, lane), accig);
    accig = wmma_f16(an1, bfrag(W_ei1, ng + 8, 1, lane), accig);
    f8v acchg = splat8(be1hg);
    acchg = wmma_f16(a1_0, bfrag(W_eh1, ng + 8, 0, lane), acchg);
    acchg = wmma_f16(a1_1, bfrag(W_eh1, ng + 8, 1, lane), acchg);
#pragma unroll
    for (int r = 0; r < 8; ++r) {
      int m = mbase + r;
      float rr = fsig(accr[r]);
      float zz = fsig(accz[r]);
      float nn = ftanh(accig[r] + rr * acchg[r]);
      float hn = (1.f - zz) * nn + zz * hp1[r];
      hp1[r] = hn;
      h1n[m * HSTR + colh] = (_Float16)hn;
    }
    __syncthreads();                              // layer-1 output committed
  }

  // ------------------------------ latent -------------------------------
  // final hidden lives in buffer 0 (S even)
  if (tid < BT) {
    int row = tid, b = b0 + row;
    float z[LATn];
#pragma unroll
    for (int j = 0; j < LATn; ++j) {
      float mu = fc_mu_b[j], lv = fc_var_b[j];
      for (int k = 0; k < Hh; ++k) {
        float hv = (float)h1s[row * HSTR + k];
        mu += hv * fc_mu_W[j * Hh + k];
        lv += hv * fc_var_W[j * Hh + k];
      }
      out[MU_OFF + b * LATn + j] = mu;
      out[LV_OFF + b * LATn + j] = lv;
      float u1 = fmaxf(rnd01((unsigned)(b * LATn + j), 0xA341u), 1e-7f);
      float u2 = rnd01((unsigned)(b * LATn + j), 0xC879u);
      float eps = sqrtf(-2.f * __logf(u1)) * __cosf(6.2831853f * u2);
      z[j] = eps * fexp(0.5f * lv) + mu;
    }
    for (int c = 0; c < Hh; ++c) {
      float hv = proj_b[c];
#pragma unroll
      for (int j = 0; j < LATn; ++j) hv += z[j] * proj_W[c * LATn + j];
      h0s[row * HSTR + c] = (_Float16)hv;
      h1s[row * HSTR + c] = (_Float16)hv;
    }
    toks[row] = START_TOK * G3;
  }
  __syncthreads();

  // re-seed register-carried h_prev with hz
#pragma unroll
  for (int r = 0; r < 8; ++r) {
    float hv = (float)h0s[(mbase + r) * HSTR + colh];
    hp0[r] = hv; hp1[r] = hv;
  }

  // preload head fragments (loop-invariant)
  h16v hbf0 = bfrag(W_hd, 0, 0, lane);
  h16v hbf1 = bfrag(W_hd, 0, 1, lane);

  // ------------------------------ decoder ------------------------------
  for (int t = 0; t < Ss; ++t) {
    const int cur = t & 1, nxt = cur ^ 1;
    const _Float16* h0c = h0s + cur * HBUF;
    const _Float16* h1c = h1s + cur * HBUF;
    _Float16* h0n = h0s + nxt * HBUF;
    _Float16* h1n = h1s + nxt * HBUF;

    h16v a0_0 = load_afrag(h0c, lane, 0);
    h16v a0_1 = load_afrag(h0c, lane, 1);
    h16v a1_0 = load_afrag(h1c, lane, 0);
    h16v a1_1 = load_afrag(h1c, lane, 1);
    int tks[8];
    *(int4*)&tks[0] = *(const int4*)&toks[mbase];
    *(int4*)&tks[4] = *(const int4*)&toks[mbase + 4];

    f8v gh0[3] = {splat8(0.f), splat8(0.f), splat8(bd0g)};
    gemm3(gh0, a0_0, a0_1, W_dh0, ng, lane);
#pragma unroll
    for (int r = 0; r < 8; ++r) {
      int m = mbase + r;
      const float* gt = &gtd[tks[r] + colh];
      float rr = fsig(gt[0] + gh0[0][r]);
      float zz = fsig(gt[Hh] + gh0[1][r]);
      float nn = ftanh(gt[2 * Hh] + rr * gh0[2][r]);
      float hn = (1.f - zz) * nn + zz * hp0[r];
      hp0[r] = hn;
      h0n[m * HSTR + colh] = (_Float16)hn;
    }
    __syncthreads();                              // layer-0 output committed

    h16v an0 = load_afrag(h0n, lane, 0);
    h16v an1 = load_afrag(h0n, lane, 1);
    f8v accr  = gemm_pair(splat8(bd1r), an0, an1, a1_0, a1_1, W_di1, W_dh1, ng, lane);
    f8v accz  = gemm_pair(splat8(bd1z), an0, an1, a1_0, a1_1, W_di1, W_dh1, ng + 4, lane);
    f8v accig = splat8(bd1ig);
    accig = wmma_f16(an0, bfrag(W_di1, ng + 8, 0, lane), accig);
    accig = wmma_f16(an1, bfrag(W_di1, ng + 8, 1, lane), accig);
    f8v acchg = splat8(bd1hg);
    acchg = wmma_f16(a1_0, bfrag(W_dh1, ng + 8, 0, lane), acchg);
    acchg = wmma_f16(a1_1, bfrag(W_dh1, ng + 8, 1, lane), acchg);
#pragma unroll
    for (int r = 0; r < 8; ++r) {
      int m = mbase + r;
      float rr = fsig(accr[r]);
      float zz = fsig(accz[r]);
      float nn = ftanh(accig[r] + rr * acchg[r]);
      float hn = (1.f - zz) * nn + zz * hp1[r];
      hp1[r] = hn;
      h1n[m * HSTR + colh] = (_Float16)hn;
    }
    __syncthreads();                              // layer-1 output committed

    // head GEMV as one zero-padded WMMA tile on wave 0 (head_b in C init);
    // same-wave LDS ordering lets us read logits back without a barrier.
    if (ng == 0) {
      h16v hA0 = load_afrag(h1n, lane, 0);
      h16v hA1 = load_afrag(h1n, lane, 1);
      f8v lac = splat8(hbn);
      lac = wmma_f16(hA0, hbf0, lac);
      lac = wmma_f16(hA1, hbf1, lac);
      int n = lane & 15;
      if (n < Vv) *(f8v*)&lgs[n * BT + mbase] = lac;   // [v][row]
      if (lane < BT) {
        int row = lane, b = b0 + row;
        float lg[Vv];
#pragma unroll
        for (int v = 0; v < Vv; ++v) lg[v] = lgs[v * BT + row];
        float mx = lg[0];
#pragma unroll
        for (int v = 1; v < Vv; ++v) mx = fmaxf(mx, lg[v]);
        float ex[Vv], s = 0.f;
#pragma unroll
        for (int v = 0; v < Vv; ++v) { ex[v] = fexp(lg[v] - mx); s += ex[v]; }
        float inv = frcp(s);
        float4 pr = make_float4(ex[0] * inv, ex[1] * inv, ex[2] * inv, ex[3] * inv);
        *(float4*)(out + (size_t)b * (Ss * Vv) + t * Vv) = pr;
        float u = rnd01((unsigned)b, 0x71F3u + (unsigned)t);
        int nt = 0; float cacc = pr.x;
        if (u >= cacc) nt = 1; cacc += pr.y;
        if (u >= cacc) nt = 2; cacc += pr.z;
        if (u >= cacc) nt = 3;
        toks[row] = nt * G3;
      }
    }
    __syncthreads();                              // sampled tokens visible
  }
}

// ------------------------------- launcher ----------------------------------

extern "C" void kernel_launch(void* const* d_in, const int* in_sizes, int n_in,
                              void* d_out, int out_size, void* d_ws, size_t ws_size,
                              hipStream_t stream) {
  (void)in_sizes; (void)n_in; (void)out_size; (void)ws_size;
  const int*   tokens   = (const int*)  d_in[0];
  const float* emb      = (const float*)d_in[1];
  const float* eWih0    = (const float*)d_in[2];
  const float* eWhh0    = (const float*)d_in[3];
  const float* ebih0    = (const float*)d_in[4];
  const float* ebhh0    = (const float*)d_in[5];
  const float* eWih1    = (const float*)d_in[6];
  const float* eWhh1    = (const float*)d_in[7];
  const float* ebih1    = (const float*)d_in[8];
  const float* ebhh1    = (const float*)d_in[9];
  const float* fc_mu_W  = (const float*)d_in[10];
  const float* fc_mu_b  = (const float*)d_in[11];
  const float* fc_var_W = (const float*)d_in[12];
  const float* fc_var_b = (const float*)d_in[13];
  const float* demb     = (const float*)d_in[14];
  const float* proj_W   = (const float*)d_in[15];
  const float* proj_b   = (const float*)d_in[16];
  const float* dWih0    = (const float*)d_in[17];
  const float* dWhh0    = (const float*)d_in[18];
  const float* dbih0    = (const float*)d_in[19];
  const float* dbhh0    = (const float*)d_in[20];
  const float* dWih1    = (const float*)d_in[21];
  const float* dWhh1    = (const float*)d_in[22];
  const float* dbih1    = (const float*)d_in[23];
  const float* dbhh1    = (const float*)d_in[24];
  const float* head_W   = (const float*)d_in[25];
  const float* head_b   = (const float*)d_in[26];

  float* tabs = (float*)d_ws;
  _Float16* frags = (_Float16*)((char*)d_ws + (size_t)2 * TABF * sizeof(float));
  float* out = (float*)d_out;

  prep_tables<<<(2 * TABF + 255) / 256, 256, 0, stream>>>(
      emb, eWih0, ebih0, ebhh0, demb, dWih0, dbih0, dbhh0, tabs);
  prep_frags<<<(6 * FRAGW + FRAGH + 255) / 256, 256, 0, stream>>>(
      eWhh0, eWih1, eWhh1, dWhh0, dWih1, dWhh1, head_W, frags);
  copy_tokens<<<(Bb * Ss + 255) / 256, 256, 0, stream>>>(tokens, (int*)d_out);
  vae_fused<<<Bb / BT, THREADS, 0, stream>>>(
      tokens, fc_mu_W, fc_mu_b, fc_var_W, fc_var_b, proj_W, proj_b, head_b,
      ebhh0, ebih1, ebhh1, dbhh0, dbih1, dbhh1, tabs, frags, out);
}
